// MultiHeadAttention_22840636080979
// MI455X (gfx1250) — compile-verified, matrix operand order checked by reading, not run on hardware
//
#include <hip/hip_runtime.h>
#include <hip/hip_bf16.h>
#include <math.h>
#include <stdint.h>

// ---------------------------------------------------------------------------
// Problem constants (from reference)
// ---------------------------------------------------------------------------
#define N_  2
#define T_  8192
#define D_  1024
#define H_  16
#define DQ_ 64
#define DV_ 64
#define S_  16
#define L_  512            // tokens per segment
#define NT_ (N_ * T_)      // 16384 rows

typedef __bf16 bf16;
typedef __attribute__((ext_vector_type(8)))  bf16  v8bf;
typedef __attribute__((ext_vector_type(16))) bf16  v16bf;
typedef __attribute__((ext_vector_type(8)))  float v8f;

// ---------------------------------------------------------------------------
// WMMA helpers (CDNA5: V_WMMA_F32_16X16X32_BF16, wave32)
// A/B fragment: 16 bf16 per lane.  lane<16: row/col = lane, K = {0..7, 16..23}
//               lane>=16: row/col = lane-16, K = {8..15, 24..31}
// C/D: 8 f32 per lane. lane<16: N=lane, M=r ; lane>=16: N=lane-16, M=8+r
// ---------------------------------------------------------------------------
__device__ __forceinline__ v8f wmma_bf16(v16bf a, v16bf b, v8f c) {
    return __builtin_amdgcn_wmma_f32_16x16x32_bf16(
        /*neg_a*/false, a, /*neg_b*/false, b,
        /*c_mod*/(short)0, c, /*reuse_a*/false, /*reuse_b*/false);
}

__device__ __forceinline__ v16bf frag_ld(const bf16* base, int stride) {
    int lane = threadIdx.x & 31;
    int r    = lane & 15;
    int kb   = (lane >> 4) << 3;          // 0 or 8
    const bf16* p = base + r * stride + kb;
    v8bf lo = *(const v8bf*)(p);
    v8bf hi = *(const v8bf*)(p + 16);
    return __builtin_shufflevector(lo, hi,
        0,1,2,3,4,5,6,7,8,9,10,11,12,13,14,15);
}

// Async copy of 16 contiguous bytes global -> LDS (gfx1250 ASYNCcnt path).
// LDS address = low 32 bits of the generic pointer (flat->LDS truncation,
// ISA 10.2).  Wait with s_wait_asynccnt before readers.
__device__ __forceinline__ void async_cp16(const bf16* gp, bf16* lp) {
    unsigned lds = (unsigned)(uintptr_t)lp;
    asm volatile("global_load_async_to_lds_b128 %0, %1, off"
                 :: "v"(lds), "v"(gp) : "memory");
}
__device__ __forceinline__ void async_wait0() {
    asm volatile("s_wait_asynccnt 0x0" ::: "memory");
}

union U16B { uint4 u; bf16 b[8]; };

// ---------------------------------------------------------------------------
// fp32 -> bf16 elementwise convert (weights)
// ---------------------------------------------------------------------------
__global__ void cvt_bf16_kernel(const float* __restrict__ src,
                                bf16* __restrict__ dst, int n) {
    int i = blockIdx.x * blockDim.x + threadIdx.x;
    if (i < n) dst[i] = (bf16)src[i];
}

// ---------------------------------------------------------------------------
// LayerNorm: one block per row of x [NT_, D_]; writes bf16 xn
// ---------------------------------------------------------------------------
__global__ void __launch_bounds__(256)
ln_kernel(const float* __restrict__ x, const float* __restrict__ g,
          const float* __restrict__ b, bf16* __restrict__ xn) {
    int row = blockIdx.x;
    const float* xr = x + (size_t)row * D_;
    __shared__ float red[256];
    float v[4];
    float s = 0.f;
    #pragma unroll
    for (int i = 0; i < 4; ++i) { v[i] = xr[threadIdx.x + i * 256]; s += v[i]; }
    red[threadIdx.x] = s; __syncthreads();
    for (int off = 128; off > 0; off >>= 1) {
        if (threadIdx.x < off) red[threadIdx.x] += red[threadIdx.x + off];
        __syncthreads();
    }
    float mu = red[0] * (1.0f / D_);
    __syncthreads();
    s = 0.f;
    #pragma unroll
    for (int i = 0; i < 4; ++i) { float d = v[i] - mu; s += d * d; }
    red[threadIdx.x] = s; __syncthreads();
    for (int off = 128; off > 0; off >>= 1) {
        if (threadIdx.x < off) red[threadIdx.x] += red[threadIdx.x + off];
        __syncthreads();
    }
    float rstd = rsqrtf(red[0] * (1.0f / D_) + 1e-5f);
    #pragma unroll
    for (int i = 0; i < 4; ++i) {
        int c = threadIdx.x + i * 256;
        xn[(size_t)row * D_ + c] = (bf16)((v[i] - mu) * rstd * g[c] + b[c]);
    }
}

// ---------------------------------------------------------------------------
// Generic WMMA GEMM: C[M,Nc] (f32) = A[M,K] bf16 @ B[K,Nc] bf16 + bias.
// Block = 256 threads = 8 waves, tile 128x128, k-step 32, double-buffered:
//   A tile staged with global_load_async_to_lds_b128 (ASYNCcnt),
//   B tile staged through registers (transpose to n-major).
// ---------------------------------------------------------------------------
#define AST 40   // LDS row pitch (elements) for A tile  (80B: 16B multiple)
#define BST 40   // LDS row pitch (elements) for B tile (n-major)

__device__ __forceinline__ void gemm_stage_A(const bf16* __restrict__ A,
                                             bf16* Asb, int mBase, int K,
                                             int k0, int tid) {
    #pragma unroll
    for (int i = 0; i < 2; ++i) {
        int c = tid + i * 256;              // 512 16B chunks: 128 rows x 4
        int row = c >> 2, q = c & 3;
        async_cp16(A + (size_t)(mBase + row) * K + k0 + q * 8,
                   Asb + row * AST + q * 8);
    }
}
__device__ __forceinline__ void gemm_load_B(const bf16* __restrict__ B,
                                            U16B* br, int nBase, int Nc,
                                            int k0, int tid) {
    #pragma unroll
    for (int i = 0; i < 2; ++i) {
        int c = tid + i * 256;              // 512 chunks: 32 k-rows x 16
        int kr = c >> 4, nq = c & 15;
        br[i].u = *(const uint4*)(B + (size_t)(k0 + kr) * Nc + nBase + nq * 8);
    }
}
__device__ __forceinline__ void gemm_store_B(bf16* Bsb, const U16B* br, int tid) {
    #pragma unroll
    for (int i = 0; i < 2; ++i) {
        int c = tid + i * 256;
        int kr = c >> 4, nq = c & 15;
        #pragma unroll
        for (int e = 0; e < 8; ++e) Bsb[(nq * 8 + e) * BST + kr] = br[i].b[e];
    }
}

__global__ void __launch_bounds__(256)
gemm_bf16_kernel(const bf16* __restrict__ A, const bf16* __restrict__ B,
                 const float* __restrict__ bias, float* __restrict__ C,
                 int M, int Nc, int K) {
    __shared__ __attribute__((aligned(16))) bf16 As[2][128 * AST];
    __shared__ __attribute__((aligned(16))) bf16 Bs[2][128 * BST];
    int tid  = threadIdx.x;
    int wid  = tid >> 5;
    int lane = tid & 31;
    int mBase = blockIdx.y * 128;
    int nBase = blockIdx.x * 128;
    int wm = wid >> 2, wn = wid & 3;

    v8f zero = {0.f,0.f,0.f,0.f,0.f,0.f,0.f,0.f};
    v8f acc[4][2];
    #pragma unroll
    for (int i = 0; i < 4; ++i)
        #pragma unroll
        for (int j = 0; j < 2; ++j) acc[i][j] = zero;

    const int NS = K >> 5;                  // k-steps of 32
    U16B breg[2];
    // prologue: stage tile 0
    gemm_stage_A(A, As[0], mBase, K, 0, tid);
    gemm_load_B(B, breg, nBase, Nc, 0, tid);
    gemm_store_B(Bs[0], breg, tid);
    async_wait0();
    __syncthreads();

    for (int ks = 0; ks < NS; ++ks) {
        int cur = ks & 1, nxt = cur ^ 1;
        bool more = (ks + 1 < NS);
        if (more) {
            gemm_stage_A(A, As[nxt], mBase, K, (ks + 1) << 5, tid);
            gemm_load_B(B, breg, nBase, Nc, (ks + 1) << 5, tid);
        }
        v16bf af[4], bfr[2];
        #pragma unroll
        for (int i = 0; i < 4; ++i)
            af[i] = frag_ld(&As[cur][(wm * 64 + i * 16) * AST], AST);
        #pragma unroll
        for (int j = 0; j < 2; ++j)
            bfr[j] = frag_ld(&Bs[cur][(wn * 32 + j * 16) * BST], BST);
        #pragma unroll
        for (int i = 0; i < 4; ++i)
            #pragma unroll
            for (int j = 0; j < 2; ++j)
                acc[i][j] = wmma_bf16(af[i], bfr[j], acc[i][j]);
        if (more) gemm_store_B(Bs[nxt], breg, tid);
        async_wait0();
        __syncthreads();
    }
    // epilogue
    #pragma unroll
    for (int i = 0; i < 4; ++i) {
        int R = mBase + wm * 64 + i * 16 + ((lane >> 4) << 3);
        #pragma unroll
        for (int j = 0; j < 2; ++j) {
            int col = nBase + wn * 32 + j * 16 + (lane & 15);
            float bv = bias[col];
            #pragma unroll
            for (int r = 0; r < 8; ++r)
                C[(size_t)(R + r) * Nc + col] = acc[i][j][r] + bv;
        }
    }
}

// ---------------------------------------------------------------------------
// RoPE + permute: src [N*T, 1024] f32 (bias already added) ->
// dst [N,H,T,64] bf16.  One thread per (n,t,h,pair).
// ---------------------------------------------------------------------------
__global__ void rope_kernel(const float* __restrict__ src, bf16* __restrict__ dst) {
    int idx = blockIdx.x * blockDim.x + threadIdx.x;   // N*T*H*32
    int i  = idx & 31;
    int t1 = idx >> 5;
    int h  = t1 & 15;
    int t2 = t1 >> 4;
    int t  = t2 & (T_ - 1);
    int n  = t2 >> 13;
    float inv = __expf(-((float)(2 * i) / 64.0f) * 9.210340371976184f); // 10000^(-2i/64)
    float fr = (float)t * inv;
    float c = cosf(fr), sn = sinf(fr);
    size_t sb = ((size_t)(n * T_ + t)) * 1024 + h * 64 + 2 * i;
    float x1 = src[sb], x2 = src[sb + 1];
    size_t db = (((size_t)(n * H_ + h)) * T_ + t) * 64 + 2 * i;
    dst[db]     = (bf16)(x1 * c - x2 * sn);
    dst[db + 1] = (bf16)(x2 * c + x1 * sn);
}

// V permute (no rope): src [N*T,1024] f32 -> dst [N,H,T,64] bf16
__global__ void vperm_kernel(const float* __restrict__ src, bf16* __restrict__ dst) {
    int idx = blockIdx.x * blockDim.x + threadIdx.x;   // N*T*H*64
    int e  = idx & 63;
    int t1 = idx >> 6;
    int h  = t1 & 15;
    int t2 = t1 >> 4;
    int t  = t2 & (T_ - 1);
    int n  = t2 >> 13;
    dst[(((size_t)(n * H_ + h)) * T_ + t) * 64 + e] =
        (bf16)src[((size_t)(n * T_ + t)) * 1024 + h * 64 + e];
}

// ---------------------------------------------------------------------------
// Memory prefix scan: one block per (n,h).  For each segment s stores the
// PREFIX memory Mem_s[64][64] = sum_{s'<s} sk^T @ v  and  u_s[64] = cumulative
// key rowsum (the degenerate z: every row of z equals u).  sk = elu(k)+1.
// ---------------------------------------------------------------------------
__global__ void __launch_bounds__(256)
memscan_kernel(const bf16* __restrict__ kb, const bf16* __restrict__ vb,
               float* __restrict__ memPre, float* __restrict__ uPre) {
    int nh = blockIdx.x;                       // 0..31
    const bf16* kp = kb + (size_t)nh * T_ * 64;
    const bf16* vp = vb + (size_t)nh * T_ * 64;
    __shared__ float sk_s[64 * 64];
    __shared__ float v_s[64 * 64];
    int tid = threadIdx.x;
    float C[16];
    #pragma unroll
    for (int i = 0; i < 16; ++i) C[i] = 0.f;
    float u = 0.f;

    for (int s = 0; s < S_; ++s) {
        float* mp = memPre + ((size_t)nh * S_ + s) * 4096;
        #pragma unroll
        for (int i = 0; i < 16; ++i) mp[tid + i * 256] = C[i];
        if (tid < 64) uPre[((size_t)nh * S_ + s) * 64 + tid] = u;
        for (int lc = 0; lc < 8; ++lc) {
            __syncthreads();
            const bf16* kc = kp + ((size_t)s * L_ + lc * 64) * 64;
            const bf16* vc = vp + ((size_t)s * L_ + lc * 64) * 64;
            #pragma unroll
            for (int i = 0; i < 16; ++i) {
                int p = tid + i * 256;
                float kv = (float)kc[p];
                sk_s[p] = (kv > 0.f ? kv : (__expf(kv) - 1.0f)) + 1.0f;
                v_s[p]  = (float)vc[p];
            }
            __syncthreads();
            #pragma unroll
            for (int i = 0; i < 16; ++i) {
                int p = tid + i * 256;
                int d = p >> 6, e = p & 63;
                float a = C[i];
                for (int l = 0; l < 64; ++l)
                    a += sk_s[l * 64 + d] * v_s[l * 64 + e];
                C[i] = a;
            }
            if (tid < 64) {
                float uu = 0.f;
                for (int l = 0; l < 64; ++l) uu += sk_s[l * 64 + tid];
                u += uu;
            }
        }
        __syncthreads();
    }
}

// ---------------------------------------------------------------------------
// Attention per (n,h,s): flash-style causal softmax over the 512-token
// segment (WMMA QK^T and PV) + linear-memory term, writes ctx bf16 in the
// reference's scrambled flattening [N][S][H][L][DV].
// K tile staged via async-to-LDS; V transposed through registers.
// ---------------------------------------------------------------------------
#define KT 72   // LDS pitch for K / Vt / P tiles (144B, 16B multiple)

__global__ void __launch_bounds__(256)
attn_kernel(const bf16* __restrict__ qb, const bf16* __restrict__ kb,
            const bf16* __restrict__ vb, const float* __restrict__ memPre,
            const float* __restrict__ uPre, const float* __restrict__ beta,
            bf16* __restrict__ ctx) {
    __shared__ __attribute__((aligned(16))) bf16 Ks[64 * KT];   // [key][d]
    __shared__ __attribute__((aligned(16))) bf16 Vt[64 * KT];   // [e][key]
    __shared__ __attribute__((aligned(16))) char Ubuf[32768];   // P(bf16)/O(f32)

    int blk = blockIdx.x;             // N*H*S = 512
    int s   = blk & 15;
    int nh  = blk >> 4;
    int n   = nh >> 4;
    int h   = nh & 15;
    int tid = threadIdx.x, wid = tid >> 5, lane = tid & 31;

    const bf16* qp = qb + ((size_t)nh * T_ + s * L_) * 64;
    const bf16* kp = kb + ((size_t)nh * T_ + s * L_) * 64;
    const bf16* vp = vb + ((size_t)nh * T_ + s * L_) * 64;
    const float* Mem = memPre + ((size_t)nh * S_ + s) * 4096;
    const float* u   = uPre   + ((size_t)nh * S_ + s) * 64;
    float g = 1.0f / (1.0f + __expf(-beta[0]));

    bf16*  Pw = (bf16*) Ubuf + wid * 16 * KT;   // per-wave 16x72 bf16
    float* Ow = (float*)Ubuf + wid * 16 * 64;   // per-wave 16x64 f32
    v8f zero = {0.f,0.f,0.f,0.f,0.f,0.f,0.f,0.f};

    for (int rb = 0; rb < 4; ++rb) {
        int rowBase = rb * 128 + wid * 16;      // segment-local rows of wave
        v16bf qf[2];
        qf[0] = frag_ld(qp + (size_t)rowBase * 64, 64);
        qf[1] = frag_ld(qp + (size_t)rowBase * 64 + 32, 64);
        v8f o[4];
        float m[8], l[8];
        #pragma unroll
        for (int i = 0; i < 4; ++i) o[i] = zero;
        #pragma unroll
        for (int r = 0; r < 8; ++r) { m[r] = -1e30f; l[r] = 0.f; }

        int jmax = rb * 2 + 1;
        for (int j = 0; j <= jmax; ++j) {
            __syncthreads();
            // K block [64][64]: async global->LDS (2 x b128 per thread)
            #pragma unroll
            for (int i = 0; i < 2; ++i) {
                int c = tid + i * 256;
                int row = c >> 3, q = c & 7;
                async_cp16(kp + ((size_t)(j * 64 + row)) * 64 + q * 8,
                           &Ks[row * KT + q * 8]);
            }
            // V block transposed [e][key] through registers
            #pragma unroll
            for (int i = 0; i < 2; ++i) {
                int c = tid + i * 256;
                int row = c >> 3, q = c & 7;
                U16B d;
                d.u = *(const uint4*)(vp + ((size_t)(j * 64 + row)) * 64 + q * 8);
                #pragma unroll
                for (int e = 0; e < 8; ++e) Vt[(q * 8 + e) * KT + row] = d.b[e];
            }
            async_wait0();
            __syncthreads();

            bool active = (rowBase + 15 >= j * 64);   // wave-uniform
            if (active) {
                v8f sc[4];
                #pragma unroll
                for (int ct = 0; ct < 4; ++ct) {
                    v8f a = zero;
                    #pragma unroll
                    for (int ks = 0; ks < 2; ++ks)
                        a = wmma_bf16(qf[ks],
                              frag_ld(&Ks[(ct * 16) * KT + ks * 32], KT), a);
                    sc[ct] = a;
                }
                int colb = j * 64 + (lane & 15);
                int rowb = rowBase + ((lane >> 4) << 3);
                #pragma unroll
                for (int ct = 0; ct < 4; ++ct)
                    #pragma unroll
                    for (int r = 0; r < 8; ++r) {
                        float sv = sc[ct][r] * 0.125f;
                        if (colb + ct * 16 > rowb + r) sv = -1e30f;
                        sc[ct][r] = sv;
                    }
                #pragma unroll
                for (int r = 0; r < 8; ++r) {
                    float mx = fmaxf(fmaxf(sc[0][r], sc[1][r]),
                                     fmaxf(sc[2][r], sc[3][r]));
                    mx = fmaxf(mx, __shfl_xor(mx, 1));
                    mx = fmaxf(mx, __shfl_xor(mx, 2));
                    mx = fmaxf(mx, __shfl_xor(mx, 4));
                    mx = fmaxf(mx, __shfl_xor(mx, 8));
                    float mn = fmaxf(m[r], mx);
                    float alpha = __expf(m[r] - mn);
                    m[r] = mn;
                    float rs = 0.f;
                    #pragma unroll
                    for (int ct = 0; ct < 4; ++ct) {
                        float p = __expf(sc[ct][r] - mn);
                        sc[ct][r] = p; rs += p;
                    }
                    rs += __shfl_xor(rs, 1); rs += __shfl_xor(rs, 2);
                    rs += __shfl_xor(rs, 4); rs += __shfl_xor(rs, 8);
                    l[r] = l[r] * alpha + rs;
                    #pragma unroll
                    for (int ct = 0; ct < 4; ++ct) o[ct][r] *= alpha;
                }
                // P -> LDS (C layout -> row-major bf16)
                #pragma unroll
                for (int ct = 0; ct < 4; ++ct)
                    #pragma unroll
                    for (int r = 0; r < 8; ++r) {
                        int rr = ((lane >> 4) << 3) + r;
                        Pw[rr * KT + ct * 16 + (lane & 15)] = (bf16)sc[ct][r];
                    }
            }
            __syncthreads();
            if (active) {
                // O += P @ V
                #pragma unroll
                for (int ks = 0; ks < 2; ++ks) {
                    v16bf a = frag_ld(&Pw[ks * 32], KT);
                    #pragma unroll
                    for (int ct = 0; ct < 4; ++ct)
                        o[ct] = wmma_bf16(a,
                                 frag_ld(&Vt[(ct * 16) * KT + ks * 32], KT),
                                 o[ct]);
                }
            }
        }
        // finalize local attention
        #pragma unroll
        for (int r = 0; r < 8; ++r) {
            float inv = 1.0f / l[r];
            #pragma unroll
            for (int ct = 0; ct < 4; ++ct) o[ct][r] *= inv;
        }
        // stage O (f32) for layout change
        #pragma unroll
        for (int ct = 0; ct < 4; ++ct)
            #pragma unroll
            for (int r = 0; r < 8; ++r) {
                int rr = ((lane >> 4) << 3) + r;
                Ow[rr * 64 + ct * 16 + (lane & 15)] = o[ct][r];
            }
        __syncthreads();
        // linear-memory term + combine.  lane -> row = lane&15, e-half = lane>>4
        {
            int row = lane & 15;
            int eb  = (lane >> 4) * 32;
            float acc[32];
            #pragma unroll
            for (int i = 0; i < 32; ++i) acc[i] = 0.f;
            float rsum = 0.f;
            const bf16* qrow = qp + (size_t)(rowBase + row) * 64;
            for (int d = 0; d < 64; ++d) {
                float qv = (float)qrow[d];
                float sq = (qv > 0.f ? qv : (__expf(qv) - 1.0f)) + 1.0f;
                rsum += sq;
                const float* mrow = Mem + d * 64 + eb;
                #pragma unroll
                for (int i = 0; i < 32; ++i) acc[i] += sq * mrow[i];
            }
            int lrow = rowBase + row;   // segment-local token
            // reference flattening: [N][S][H][L][DV]
            bf16* crow = ctx + (size_t)n * T_ * 1024
                             + (((size_t)s * H_ + h) * L_ + lrow) * 64 + eb;
            #pragma unroll
            for (int i = 0; i < 32; ++i) {
                float am  = acc[i] / (rsum * u[eb + i] + 1e-6f);
                float loc = Ow[row * 64 + eb + i];
                crow[i] = (bf16)(g * am + (1.0f - g) * loc);
            }
        }
        __syncthreads();
    }
}

// ---------------------------------------------------------------------------
// Host-side orchestration
// ---------------------------------------------------------------------------
extern "C" void kernel_launch(void* const* d_in, const int* in_sizes, int n_in,
                              void* d_out, int out_size, void* d_ws, size_t ws_size,
                              hipStream_t stream) {
    const float* x    = (const float*)d_in[0];
    const float* ln_g = (const float*)d_in[1];
    const float* ln_b = (const float*)d_in[2];
    const float* Wq   = (const float*)d_in[3];
    const float* bq   = (const float*)d_in[4];
    const float* Wk   = (const float*)d_in[5];
    const float* bk   = (const float*)d_in[6];
    const float* Wv   = (const float*)d_in[7];
    const float* bv   = (const float*)d_in[8];
    const float* Wo   = (const float*)d_in[9];
    const float* bo   = (const float*)d_in[10];
    const float* beta = (const float*)d_in[11];
    float* out = (float*)d_out;

    char* ws = (char*)d_ws;
    size_t off = 0;
    auto alloc = [&](size_t bytes) -> void* {
        void* p = ws + off;
        off += (bytes + 255) & ~(size_t)255;
        return p;
    };
    bf16*  xn     = (bf16*) alloc((size_t)NT_ * D_ * sizeof(bf16));
    bf16*  Wqb    = (bf16*) alloc((size_t)D_ * D_ * sizeof(bf16));
    bf16*  Wkb    = (bf16*) alloc((size_t)D_ * D_ * sizeof(bf16));
    bf16*  Wvb    = (bf16*) alloc((size_t)D_ * D_ * sizeof(bf16));
    bf16*  Wob    = (bf16*) alloc((size_t)D_ * D_ * sizeof(bf16));
    float* stage  = (float*)alloc((size_t)NT_ * D_ * sizeof(float)); // reused q/k/v
    bf16*  qbuf   = (bf16*) alloc((size_t)N_ * H_ * T_ * 64 * sizeof(bf16));
    bf16*  kbuf   = (bf16*) alloc((size_t)N_ * H_ * T_ * 64 * sizeof(bf16));
    bf16*  vbuf   = (bf16*) alloc((size_t)N_ * H_ * T_ * 64 * sizeof(bf16));
    float* memPre = (float*)alloc((size_t)N_ * H_ * S_ * 64 * 64 * sizeof(float));
    float* uPre   = (float*)alloc((size_t)N_ * H_ * S_ * 64 * sizeof(float));
    bf16*  ctx    = (bf16*) alloc((size_t)NT_ * D_ * sizeof(bf16));
    (void)ws_size; (void)n_in; (void)in_sizes; (void)out_size;

    const int wn = D_ * D_;                 // 1M weight elements
    cvt_bf16_kernel<<<wn / 256, 256, 0, stream>>>(Wq, Wqb, wn);
    cvt_bf16_kernel<<<wn / 256, 256, 0, stream>>>(Wk, Wkb, wn);
    cvt_bf16_kernel<<<wn / 256, 256, 0, stream>>>(Wv, Wvb, wn);
    cvt_bf16_kernel<<<wn / 256, 256, 0, stream>>>(Wo, Wob, wn);

    ln_kernel<<<NT_, 256, 0, stream>>>(x, ln_g, ln_b, xn);

    dim3 ggrid(D_ / 128, NT_ / 128);        // (8, 128)
    const int ropeN  = N_ * T_ * H_ * 32;   // pairs
    const int vpermN = N_ * T_ * H_ * 64;

    // Q projection + rope
    gemm_bf16_kernel<<<ggrid, 256, 0, stream>>>(xn, Wqb, bq, stage, NT_, D_, D_);
    rope_kernel<<<ropeN / 256, 256, 0, stream>>>(stage, qbuf);
    // K projection + rope
    gemm_bf16_kernel<<<ggrid, 256, 0, stream>>>(xn, Wkb, bk, stage, NT_, D_, D_);
    rope_kernel<<<ropeN / 256, 256, 0, stream>>>(stage, kbuf);
    // V projection + permute
    gemm_bf16_kernel<<<ggrid, 256, 0, stream>>>(xn, Wvb, bv, stage, NT_, D_, D_);
    vperm_kernel<<<vpermN / 256, 256, 0, stream>>>(stage, vbuf);

    // memory prefix states
    memscan_kernel<<<N_ * H_, 256, 0, stream>>>(kbuf, vbuf, memPre, uPre);

    // attention (all (n,h,segment) blocks in parallel)
    attn_kernel<<<N_ * H_ * S_, 256, 0, stream>>>(qbuf, kbuf, vbuf,
                                                  memPre, uPre, beta, ctx);

    // output projection
    gemm_bf16_kernel<<<ggrid, 256, 0, stream>>>(ctx, Wob, bo, out, NT_, D_, D_);
}
